// FC_STGNN_SSC_56470230008618
// MI455X (gfx1250) — matrix operand order repaired and checked by reading, not compiled, not run on hardware
//
#include <hip/hip_runtime.h>
#include <hip/hip_bf16.h>

// ---------------------------------------------------------------------------
// FC-STGNN forward for MI455X (gfx1250, wave32, WMMA).
// Big GEMMs: 32x32 tile per wave, 4 independent v_wmma_f32_16x16x32_bf16 per
// K-step, ping-pong double-buffered b128 fragment loads in fixed registers
// (no rotation movs, no per-iteration address recomputation).
// ---------------------------------------------------------------------------

typedef __bf16 bf16;
typedef __attribute__((ext_vector_type(8)))  bf16  v8bf;
typedef __attribute__((ext_vector_type(16))) bf16  v16bf;
typedef __attribute__((ext_vector_type(8)))  float v8f;

union Frag16 { v16bf v; v8bf h[2]; };

__device__ inline v8f wmma_bf16(const Frag16& a, const Frag16& b, v8f c) {
  // 8 args: (neg_a, A, neg_b, B, c_mod, C, reuse_a, reuse_b)
  return __builtin_amdgcn_wmma_f32_16x16x32_bf16(false, a.v, false, b.v,
                                                 (short)0, c, false, false);
}

// A fragment: lane<16 holds K {0..7,16..23}, lane>=16 holds K {8..15,24..31}
__device__ inline void loadA(Frag16& f, const bf16* p, int offA) {
  f.h[0] = *(const v8bf*)(p + offA);
  f.h[1] = *(const v8bf*)(p + 16 + offA);
}
// B fragment: lane<16 holds K {0..7,16..23}, lane>=16 holds K {8..15,24..31}
__device__ inline void loadB(Frag16& f, const bf16* p, int offB) {
  f.h[0] = *(const v8bf*)(p + (offB >> 1));
  f.h[1] = *(const v8bf*)(p + 16 + (offB >> 1));
}

// ---------------------------------------------------------------------------
// Dense GEMM, 32x32 tile per wave (block = 128 thr = 4 waves = 64x64 tile).
// C[M,N] = A[M,K] * Bt[N,K]^T. M % 64 == 0, N = gridDim.y*64, K % 64 == 0.
// mode 0: outF = acc + bias
// mode 1: outF = BN(acc + bias)
// mode 3: outF = leaky_relu(BN(acc + bias))
// bn layout: [gamma | beta | mean | var], each length N.
// ---------------------------------------------------------------------------
__global__ void gemm32(const bf16* __restrict__ A, const bf16* __restrict__ Bt,
                       const float* __restrict__ bias, const float* __restrict__ bn,
                       float* __restrict__ outF, int M, int N, int K, int mode)
{
  const int lane = threadIdx.x & 31;
  const int wave = threadIdx.x >> 5;
  const int mw   = wave & 1;
  const int nw2  = wave >> 1;
  const int rowBase = blockIdx.x * 64 + mw * 32;
  const int colBase = blockIdx.y * 64 + nw2 * 32;
  const int rsub = lane & 15;
  const int offA = (lane >> 4) * 8;
  const int offB = (lane >> 4) * 16;

  const bf16* pa0 = A  + (size_t)(rowBase + rsub) * K;
  const bf16* pa1 = pa0 + (size_t)16 * K;
  const bf16* pb0 = Bt + (size_t)(colBase + rsub) * K;
  const bf16* pb1 = pb0 + (size_t)16 * K;

  v8f acc00 = {}, acc01 = {}, acc10 = {}, acc11 = {};
  Frag16 a0f[2], a1f[2], b0f[2], b1f[2];

  loadA(a0f[0], pa0, offA);  loadA(a1f[0], pa1, offA);
  loadB(b0f[0], pb0, offB);  loadB(b1f[0], pb1, offB);

  for (int k = 0; k < K; k += 64) {
    // phase-1 fragments (k+32)
    loadA(a0f[1], pa0 + 32, offA);  loadA(a1f[1], pa1 + 32, offA);
    loadB(b0f[1], pb0 + 32, offB);  loadB(b1f[1], pb1 + 32, offB);
    // 4 independent WMMAs on phase-0
    acc00 = wmma_bf16(a0f[0], b0f[0], acc00);
    acc01 = wmma_bf16(a0f[0], b1f[0], acc01);
    acc10 = wmma_bf16(a1f[0], b0f[0], acc10);
    acc11 = wmma_bf16(a1f[0], b1f[0], acc11);
    pa0 += 64; pa1 += 64; pb0 += 64; pb1 += 64;
    if (k + 64 < K) {                  // phase-0 fragments for next iteration
      loadA(a0f[0], pa0, offA);  loadA(a1f[0], pa1, offA);
      loadB(b0f[0], pb0, offB);  loadB(b1f[0], pb1, offB);
    }
    // 4 independent WMMAs on phase-1
    acc00 = wmma_bf16(a0f[1], b0f[1], acc00);
    acc01 = wmma_bf16(a0f[1], b1f[1], acc01);
    acc10 = wmma_bf16(a1f[1], b0f[1], acc10);
    acc11 = wmma_bf16(a1f[1], b1f[1], acc11);
  }

  v8f accs[2][2] = { { acc00, acc01 }, { acc10, acc11 } };
  const int rB = rowBase + ((lane >> 4) * 8);
  const int cB = colBase + (lane & 15);
#pragma unroll
  for (int sn = 0; sn < 2; sn++) {
    const int c = cB + sn * 16;
    const float bs = bias[c];
    float g = 0.f, be = 0.f, mu = 0.f, va = 1.f;
    if (mode == 1 || mode == 3) {
      g = bn[c]; be = bn[N + c]; mu = bn[2 * N + c]; va = bn[3 * N + c];
    }
    const float sc = g * rsqrtf(va + 1e-5f);
#pragma unroll
    for (int sm = 0; sm < 2; sm++) {
#pragma unroll
      for (int i = 0; i < 8; i++) {
        const int r = rB + sm * 16 + i;
        float f = accs[sm][sn][i] + bs;
        if (mode == 0) {
          outF[(size_t)r * N + c] = f;
        } else if (mode == 1) {
          outF[(size_t)r * N + c] = (f - mu) * sc + be;
        } else {
          float t = (f - mu) * sc + be;
          outF[(size_t)r * N + c] = t > 0.f ? t : 0.01f * t;
        }
      }
    }
  }
}

// ---------------------------------------------------------------------------
// Small dense GEMM (16x16 tile per wave) for the M=32 FC head.
// outB = bf16(relu(acc + bias))
// ---------------------------------------------------------------------------
__global__ void gemm16(const bf16* __restrict__ A, const bf16* __restrict__ Bt,
                       const float* __restrict__ bias, bf16* __restrict__ outB,
                       int M, int N, int K)
{
  const int lane = threadIdx.x & 31;
  const int wave = threadIdx.x >> 5;
  const int mt   = blockIdx.x;
  const int nt   = blockIdx.y * 4 + wave;
  const int row  = mt * 16 + (lane & 15);
  const int col  = nt * 16 + (lane & 15);
  const int offA = (lane >> 4) * 8;
  const int offB = (lane >> 4) * 16;

  const bf16* aRow = A  + (size_t)row * K;
  const bf16* bCol = Bt + (size_t)col * K;

  v8f acc = {};
  for (int k = 0; k < K; k += 32) {
    Frag16 af, bfv;
    loadA(af, aRow + k, offA);
    loadB(bfv, bCol + k, offB);
    acc = wmma_bf16(af, bfv, acc);
  }
  const float bs = bias[col];
#pragma unroll
  for (int i = 0; i < 8; i++) {
    const int r = mt * 16 + i + ((lane >> 4) * 8);
    outB[(size_t)r * N + col] = (bf16)fmaxf(acc[i] + bs, 0.f);
  }
}

// ---------------------------------------------------------------------------
// Conv1d(k=3, pad) as 3 shifted implicit GEMMs, 32x32 tile per wave.
// X per seq: [Lin, Cin] bf16 (pos-major), Wt: [3][Cout][Cin] bf16,
// Y per seq: [Lout, Cout] bf16 with BN+ReLU fused.
// OOB rows are data-masked to zero (EXEC stays all-ones for WMMA).
// Cin % 64 == 0 at both call sites (128, 256).
// ---------------------------------------------------------------------------
__global__ void conv_gemm(const bf16* __restrict__ X, const bf16* __restrict__ Wt,
                          const float* __restrict__ bn, bf16* __restrict__ Y,
                          int Lin, int Cin, int Lout, int Cout, int pad)
{
  const int lane = threadIdx.x & 31;
  const int wave = threadIdx.x >> 5;
  const int mw   = wave & 1;
  const int nw2  = wave >> 1;
  const int s    = blockIdx.z;
  const int rowBase = blockIdx.x * 64 + mw * 32;
  const int colBase = blockIdx.y * 64 + nw2 * 32;
  const int rsub = lane & 15;
  const int offA = (lane >> 4) * 8;
  const int offB = (lane >> 4) * 16;
  const int row0 = rowBase + rsub;        // A stream 0
  const int row1 = row0 + 16;             // A stream 1

  const bf16* xS = X + (size_t)s * Lin * Cin;
  const bf16* wCol0 = Wt + (size_t)(colBase + rsub) * Cin;
  const bf16* wCol1 = wCol0 + (size_t)16 * Cin;

  Frag16 zf;
#pragma unroll
  for (int e = 0; e < 16; e++) zf.v[e] = (bf16)0.0f;

  v8f acc00 = {}, acc01 = {}, acc10 = {}, acc11 = {};
  Frag16 a0f[2], a1f[2], b0f[2], b1f[2];

  for (int sh = 0; sh < 3; sh++) {
    const int li0 = row0 - pad + sh;
    const int li1 = row1 - pad + sh;
    const bool ok0 = (li0 >= 0) && (li0 < Lin);
    const bool ok1 = (li1 >= 0) && (li1 < Lin);
    const bf16* pa0 = xS + (size_t)(ok0 ? li0 : 0) * Cin;
    const bf16* pa1 = xS + (size_t)(ok1 ? li1 : 0) * Cin;
    const bf16* pb0 = wCol0 + (size_t)sh * Cout * Cin;
    const bf16* pb1 = wCol1 + (size_t)sh * Cout * Cin;

    if (ok0) loadA(a0f[0], pa0, offA); else a0f[0] = zf;
    if (ok1) loadA(a1f[0], pa1, offA); else a1f[0] = zf;
    loadB(b0f[0], pb0, offB);
    loadB(b1f[0], pb1, offB);

    for (int k = 0; k < Cin; k += 64) {
      if (ok0) loadA(a0f[1], pa0 + 32, offA); else a0f[1] = zf;
      if (ok1) loadA(a1f[1], pa1 + 32, offA); else a1f[1] = zf;
      loadB(b0f[1], pb0 + 32, offB);
      loadB(b1f[1], pb1 + 32, offB);
      acc00 = wmma_bf16(a0f[0], b0f[0], acc00);
      acc01 = wmma_bf16(a0f[0], b1f[0], acc01);
      acc10 = wmma_bf16(a1f[0], b0f[0], acc10);
      acc11 = wmma_bf16(a1f[0], b1f[0], acc11);
      pa0 += 64; pa1 += 64; pb0 += 64; pb1 += 64;
      if (k + 64 < Cin) {
        if (ok0) loadA(a0f[0], pa0, offA); else a0f[0] = zf;
        if (ok1) loadA(a1f[0], pa1, offA); else a1f[0] = zf;
        loadB(b0f[0], pb0, offB);
        loadB(b1f[0], pb1, offB);
      }
      acc00 = wmma_bf16(a0f[1], b0f[1], acc00);
      acc01 = wmma_bf16(a0f[1], b1f[1], acc01);
      acc10 = wmma_bf16(a1f[1], b0f[1], acc10);
      acc11 = wmma_bf16(a1f[1], b1f[1], acc11);
    }
  }

  v8f accs[2][2] = { { acc00, acc01 }, { acc10, acc11 } };
  const int rB = rowBase + ((lane >> 4) * 8);
  const int cB = colBase + (lane & 15);
#pragma unroll
  for (int sn = 0; sn < 2; sn++) {
    const int c = cB + sn * 16;
    const float sc = bn[c] * rsqrtf(bn[3 * Cout + c] + 1e-5f);
    const float be = bn[Cout + c], mu = bn[2 * Cout + c];
#pragma unroll
    for (int sm = 0; sm < 2; sm++) {
#pragma unroll
      for (int i = 0; i < 8; i++) {
        const int r = rB + sm * 16 + i;
        if (r < Lout) {
          float f = (accs[sm][sn][i] - mu) * sc + be;
          Y[((size_t)s * Lout + r) * Cout + c] = (bf16)fmaxf(f, 0.f);
        }
      }
    }
  }
}

// ---------------------------------------------------------------------------
// conv1 (Cin=1) + BN + ReLU + MaxPool(k2,s2,pad1) fused. Out: [768,751,128] bf16
// ---------------------------------------------------------------------------
__global__ void conv1_pool(const float* __restrict__ X, const float* __restrict__ W,
                           const float* __restrict__ bn, bf16* __restrict__ Y)
{
  const long long total = 768LL * 751 * 128;
  long long idx = (long long)blockIdx.x * blockDim.x + threadIdx.x;
  if (idx >= total) return;
  const int o = (int)(idx & 127);
  const int p = (int)((idx >> 7) % 751);
  const int s = (int)(idx / (751 * 128));
  const float w0 = W[o * 3 + 0], w1 = W[o * 3 + 1], w2 = W[o * 3 + 2];
  const float sc = bn[o] * rsqrtf(bn[384 + o] + 1e-5f);
  const float be = bn[128 + o], mu = bn[256 + o];
  const float* xs = X + (size_t)s * 1500;
  float best = -1e30f;
#pragma unroll
  for (int q = 0; q < 2; q++) {
    const int l = 2 * p - 1 + q;
    if (l < 0 || l >= 1500) continue;
    float a = w1 * xs[l];
    if (l - 1 >= 0)   a += w0 * xs[l - 1];
    if (l + 1 < 1500) a += w2 * xs[l + 1];
    best = fmaxf(best, fmaxf((a - mu) * sc + be, 0.f));
  }
  Y[idx] = (bf16)best;
}

// MaxPool1d(k=2, s=2, pad=1) on [S, Lc, C] bf16 -> [S, Lp, C] bf16
__global__ void maxpool2(const bf16* __restrict__ in, bf16* __restrict__ out,
                         int S, int Lc, int Lp, int C)
{
  long long total = (long long)S * Lp * C;
  long long idx = (long long)blockIdx.x * blockDim.x + threadIdx.x;
  if (idx >= total) return;
  const int c = (int)(idx % C);
  const int p = (int)((idx / C) % Lp);
  const int s = (int)(idx / ((long long)C * Lp));
  const int l0 = 2 * p - 1, l1 = 2 * p;
  float best = -1e30f;
  if (l0 >= 0) best = (float)in[((size_t)s * Lc + l0) * C + c];
  if (l1 < Lc) best = fmaxf(best, (float)in[((size_t)s * Lc + l1) * C + c]);
  out[idx] = (bf16)best;
}

// Positional encoding add on map2 output a[768,256] (f32)
__global__ void add_pe(float* __restrict__ a)
{
  int idx = blockIdx.x * blockDim.x + threadIdx.x;   // 768*256
  const int d = idx & 255;
  const int s = idx >> 8;
  const int t = (s >> 2) % 6;
  const int i2 = d & ~1;
  const float div = __expf(-(float)i2 * 0.017989255f);  // ln(100)/256
  const float ang = (float)t * div;
  a[idx] += (d & 1) ? __cosf(ang) : __sinf(ang);
}

// ------------------------- weight prepack kernels --------------------------
__global__ void pack_conv_w(const float* __restrict__ w, bf16* __restrict__ wt,
                            int O, int I)
{
  long long total = (long long)O * I * 3;
  long long idx = (long long)blockIdx.x * blockDim.x + threadIdx.x;
  if (idx >= total) return;
  const int k = (int)(idx % 3);
  const int i = (int)((idx / 3) % I);
  const int o = (int)(idx / (3LL * I));
  wt[((size_t)k * O + o) * I + i] = (bf16)w[idx];   // w[o][i][k]
}

__global__ void pack_lin(const float* __restrict__ w, bf16* __restrict__ wt,
                         int K, int N)
{
  long long total = (long long)K * N;
  long long idx = (long long)blockIdx.x * blockDim.x + threadIdx.x;
  if (idx >= total) return;
  const int n = (int)(idx % N);
  const long long k = idx / N;
  wt[(size_t)n * K + k] = (bf16)w[idx];             // w[k][n] -> wt[n][k]
}

// map2_w[c*191+l][n] -> wmap[n][l*128+c]  (activations stay position-major)
__global__ void pack_map2(const float* __restrict__ w, bf16* __restrict__ wt)
{
  long long total = 24448LL * 256;
  long long idx = (long long)blockIdx.x * blockDim.x + threadIdx.x;
  if (idx >= total) return;
  const int n  = (int)(idx % 256);
  const int kk = (int)(idx / 256);        // kk = c*191 + l
  const int c = kk / 191, l = kk % 191;
  wt[(size_t)n * 24448 + (size_t)l * 128 + c] = (bf16)w[idx];
}

// ----------------------------- MPNN kernels --------------------------------
// Gather window nodes into packed Xc (raw bf16) and Xb (BN'd bf16).
__global__ void pack_xc(const float* __restrict__ aF, const float* __restrict__ bnA,
                        bf16* __restrict__ Xc, bf16* __restrict__ Xb,
                        int Bp, int nw, int stride)
{
  long long total = (long long)Bp * 2048;
  long long idx = (long long)blockIdx.x * blockDim.x + threadIdx.x;
  if (idx >= total) return;
  const int d  = (int)(idx & 255);
  const int n  = (int)((idx >> 8) & 7);
  const int bi = (int)(idx >> 11);
  const int b = bi / nw, j = bi % nw;
  const int t = j * stride + (n >> 2), sensor = n & 3;
  const int s = (b * 6 + t) * 4 + sensor;
  const float v = aF[(size_t)s * 256 + d];
  Xc[idx] = (bf16)v;
  const float sc = bnA[d] * rsqrtf(bnA[768 + d] + 1e-5f);
  Xb[idx] = (bf16)((v - bnA[512 + d]) * sc + bnA[256 + d]);
}

// Per-window 8x8 adjacency: leaky(f f^T - eye*1e8) -> softmax -> +eye -> decay
__global__ void adj_kernel(const float* __restrict__ f, float* __restrict__ Adj)
{
  __shared__ float sf[8 * 256];
  __shared__ float sA[64];
  const int bi = blockIdx.x;
  const int tid = threadIdx.x;                 // 64 threads
  for (int i = tid; i < 2048; i += 64) sf[i] = f[(size_t)bi * 2048 + i];
  __syncthreads();
  const int n = tid >> 3, m = tid & 7;
  float dot = 0.f;
  for (int d = 0; d < 256; d++) dot += sf[n * 256 + d] * sf[m * 256 + d];
  float v = dot - (n == m ? 1e8f : 0.f);
  v = v > 0.f ? v : 0.01f * v;
  sA[tid] = v;
  __syncthreads();
  float mx = -1e30f;
  for (int q = 0; q < 8; q++) mx = fmaxf(mx, sA[n * 8 + q]);
  float sum = 0.f;
  for (int q = 0; q < 8; q++) sum += __expf(sA[n * 8 + q] - mx);
  float val = __expf(v - mx) / sum + (n == m ? 1.f : 0.f);
  const float mask = ((n >> 2) == (m >> 2)) ? 1.f : 0.7f;   // DECAY^|dt|
  Adj[(size_t)bi * 64 + tid] = val * mask;
}

// y = Adj @ Xb  (8x8 x 8x256 per window), out bf16 packed for next GEMM
__global__ void adjx(const float* __restrict__ Adj, const bf16* __restrict__ Xb,
                     bf16* __restrict__ Y, int Bp)
{
  long long total = (long long)Bp * 2048;
  long long idx = (long long)blockIdx.x * blockDim.x + threadIdx.x;
  if (idx >= total) return;
  const int d  = (int)(idx & 255);
  const int n  = (int)((idx >> 8) & 7);
  const int bi = (int)(idx >> 11);
  float acc = 0.f;
#pragma unroll
  for (int m = 0; m < 8; m++)
    acc += Adj[(size_t)bi * 64 + n * 8 + m] * (float)Xb[((size_t)bi * 8 + m) * 256 + d];
  Y[idx] = (bf16)acc;
}

// mean over window-time + scatter into feats[32, 4096]
__global__ void mean_scatter(const float* __restrict__ H, bf16* __restrict__ feats,
                             int Bp, int nw, int off)
{
  long long total = (long long)Bp * 512;
  long long idx = (long long)blockIdx.x * blockDim.x + threadIdx.x;
  if (idx >= total) return;
  const int d = (int)(idx & 127);
  const int sensor = (int)((idx >> 7) & 3);
  const int bi = (int)(idx >> 9);
  const float v = 0.5f * (H[((size_t)bi * 8 + sensor) * 128 + d] +
                          H[((size_t)bi * 8 + 4 + sensor) * 128 + d]);
  const int b = bi / nw, j = bi % nw;
  feats[(size_t)b * 4096 + off + j * 512 + sensor * 128 + d] = (bf16)v;
}

// final 128 -> 5 linear, f32 output
__global__ void fc4_kernel(const bf16* __restrict__ h, const float* __restrict__ w,
                           const float* __restrict__ b, float* __restrict__ out)
{
  const int idx = threadIdx.x;              // 160
  if (idx >= 160) return;
  const int j = idx % 5, bb = idx / 5;
  float acc = b[j];
  for (int k = 0; k < 128; k++) acc += (float)h[bb * 128 + k] * w[k * 5 + j];
  out[idx] = acc;
}

// ---------------------------------------------------------------------------
extern "C" void kernel_launch(void* const* d_in, const int* in_sizes, int n_in,
                              void* d_out, int out_size, void* d_ws, size_t ws_size,
                              hipStream_t stream)
{
  (void)in_sizes; (void)n_in; (void)out_size; (void)ws_size;

  const float* X       = (const float*)d_in[0];
  const float* conv1_w = (const float*)d_in[1];
  const float* bn1     = (const float*)d_in[2];
  const float* conv2_w = (const float*)d_in[3];
  const float* bn2     = (const float*)d_in[4];
  const float* conv3_w = (const float*)d_in[5];
  const float* bn3     = (const float*)d_in[6];
  const float* map2_w  = (const float*)d_in[7];
  const float* map2_b  = (const float*)d_in[8];
  const float* bn_map2 = (const float*)d_in[9];
  const float* g1_w    = (const float*)d_in[10];
  const float* g1_b    = (const float*)d_in[11];
  const float* bnA1    = (const float*)d_in[12];
  const float* th1_w   = (const float*)d_in[13];
  const float* th1_b   = (const float*)d_in[14];
  const float* bnM1    = (const float*)d_in[15];
  const float* g2_w    = (const float*)d_in[16];
  const float* g2_b    = (const float*)d_in[17];
  const float* bnA2    = (const float*)d_in[18];
  const float* th2_w   = (const float*)d_in[19];
  const float* th2_b   = (const float*)d_in[20];
  const float* bnM2    = (const float*)d_in[21];
  const float* fc1_w   = (const float*)d_in[22];
  const float* fc1_b   = (const float*)d_in[23];
  const float* fc2_w   = (const float*)d_in[24];
  const float* fc2_b   = (const float*)d_in[25];
  const float* fc3_w   = (const float*)d_in[26];
  const float* fc3_b   = (const float*)d_in[27];
  const float* fc4_w   = (const float*)d_in[28];
  const float* fc4_b   = (const float*)d_in[29];

  // ---- workspace slab allocation (peak ~466 MB, with region aliasing) ----
  char* w = (char*)d_ws;
  size_t off = 0;
  auto alloc = [&](size_t bytes) -> char* {
    char* p = w + off;
    off += (bytes + 255) & ~(size_t)255;
    return p;
  };

  // R1: a1 [768,751,128]bf16 first; later reused for pooled conv2 [768,377,256]bf16
  char* R1 = alloc(148242432);
  // R2: conv2 raw [768,753,256]bf16; later reused for conv3 raw + pooled conv3
  char* R2 = alloc(296091648);

  bf16* a1     = (bf16*)R1;
  bf16* c2pool = (bf16*)R1;
  bf16* c2raw  = (bf16*)R2;
  bf16* c3raw  = (bf16*)R2;                     // 74,907,648 B
  bf16* c3pool = (bf16*)(R2 + 74907648);        // 37,552,128 B

  bf16* wt2  = (bf16*)alloc(3 * 256 * 128 * 2);
  bf16* wt3  = (bf16*)alloc(3 * 128 * 256 * 2);
  bf16* wmap = (bf16*)alloc((size_t)24448 * 256 * 2);
  bf16* g1t  = (bf16*)alloc(256 * 256 * 2);
  bf16* g2t  = (bf16*)alloc(256 * 256 * 2);
  bf16* th1t = (bf16*)alloc(128 * 256 * 2);
  bf16* th2t = (bf16*)alloc(128 * 256 * 2);
  bf16* fc1t = (bf16*)alloc((size_t)4096 * 256 * 2);
  bf16* fc2t = (bf16*)alloc(256 * 256 * 2);
  bf16* fc3t = (bf16*)alloc(256 * 128 * 2);

  float* aF  = (float*)alloc(768 * 256 * 4);
  bf16* Xc1  = (bf16*)alloc(1280 * 256 * 2);
  bf16* Xb1  = (bf16*)alloc(1280 * 256 * 2);
  float* f1  = (float*)alloc(1280 * 256 * 4);
  float* A1  = (float*)alloc(160 * 64 * 4);
  bf16* y1   = (bf16*)alloc(1280 * 256 * 2);
  float* h1  = (float*)alloc(1280 * 128 * 4);
  bf16* Xc2  = (bf16*)alloc(768 * 256 * 2);
  bf16* Xb2  = (bf16*)alloc(768 * 256 * 2);
  float* f2  = (float*)alloc(768 * 256 * 4);
  float* A2  = (float*)alloc(96 * 64 * 4);
  bf16* y2   = (bf16*)alloc(768 * 256 * 2);
  float* h2  = (float*)alloc(768 * 128 * 4);
  bf16* feats = (bf16*)alloc(32 * 4096 * 2);
  bf16* hf1  = (bf16*)alloc(32 * 256 * 2);
  bf16* hf2  = (bf16*)alloc(32 * 256 * 2);
  bf16* hf3  = (bf16*)alloc(32 * 128 * 2);

  auto cdiv = [](long long a, long long b) { return (unsigned)((a + b - 1) / b); };

  // ---- weight prepack (bf16, B-transposed layouts) ----
  pack_conv_w<<<cdiv(256LL * 128 * 3, 256), 256, 0, stream>>>(conv2_w, wt2, 256, 128);
  pack_conv_w<<<cdiv(128LL * 256 * 3, 256), 256, 0, stream>>>(conv3_w, wt3, 128, 256);
  pack_map2 <<<cdiv(24448LL * 256, 256), 256, 0, stream>>>(map2_w, wmap);
  pack_lin  <<<cdiv(256LL * 256, 256),  256, 0, stream>>>(g1_w,  g1t,  256, 256);
  pack_lin  <<<cdiv(256LL * 256, 256),  256, 0, stream>>>(g2_w,  g2t,  256, 256);
  pack_lin  <<<cdiv(256LL * 128, 256),  256, 0, stream>>>(th1_w, th1t, 256, 128);
  pack_lin  <<<cdiv(256LL * 128, 256),  256, 0, stream>>>(th2_w, th2t, 256, 128);
  pack_lin  <<<cdiv(4096LL * 256, 256), 256, 0, stream>>>(fc1_w, fc1t, 4096, 256);
  pack_lin  <<<cdiv(256LL * 256, 256),  256, 0, stream>>>(fc2_w, fc2t, 256, 256);
  pack_lin  <<<cdiv(256LL * 128, 256),  256, 0, stream>>>(fc3_w, fc3t, 256, 128);

  // ---- conv stack (convs 2/3 as WMMA implicit GEMMs, 64x64 block tiles) ----
  conv1_pool<<<cdiv(768LL * 751 * 128, 256), 256, 0, stream>>>(X, conv1_w, bn1, a1);

  conv_gemm<<<dim3(12, 4, 768), 128, 0, stream>>>(a1, wt2, bn2, c2raw,
                                                  751, 128, 753, 256, 2);
  maxpool2 <<<cdiv(768LL * 377 * 256, 256), 256, 0, stream>>>(c2raw, c2pool,
                                                              768, 753, 377, 256);
  conv_gemm<<<dim3(6, 2, 768), 128, 0, stream>>>(c2pool, wt3, bn3, c3raw,
                                                 377, 256, 381, 128, 3);
  maxpool2 <<<cdiv(768LL * 191 * 128, 256), 256, 0, stream>>>(c3raw, c3pool,
                                                              768, 381, 191, 128);

  // ---- map2 + BN + positional encoding ----
  gemm32<<<dim3(12, 4), 128, 0, stream>>>(c3pool, wmap, map2_b, bn_map2,
                                          aF, 768, 256, 24448, 1);
  add_pe<<<768, 256, 0, stream>>>(aF);

  // ---- MPNN block 1 (win=2, stride=1, nw=5, B'=160) ----
  pack_xc<<<cdiv(160LL * 2048, 256), 256, 0, stream>>>(aF, bnA1, Xc1, Xb1, 160, 5, 1);
  gemm32 <<<dim3(20, 4), 128, 0, stream>>>(Xc1, g1t, g1_b, nullptr, f1,
                                           1280, 256, 256, 0);
  adj_kernel<<<160, 64, 0, stream>>>(f1, A1);
  adjx  <<<cdiv(160LL * 2048, 256), 256, 0, stream>>>(A1, Xb1, y1, 160);
  gemm32<<<dim3(20, 2), 128, 0, stream>>>(y1, th1t, th1_b, bnM1, h1,
                                          1280, 128, 256, 3);
  mean_scatter<<<cdiv(160LL * 512, 256), 256, 0, stream>>>(h1, feats, 160, 5, 0);

  // ---- MPNN block 2 (win=2, stride=2, nw=3, B'=96) ----
  pack_xc<<<cdiv(96LL * 2048, 256), 256, 0, stream>>>(aF, bnA2, Xc2, Xb2, 96, 3, 2);
  gemm32 <<<dim3(12, 4), 128, 0, stream>>>(Xc2, g2t, g2_b, nullptr, f2,
                                           768, 256, 256, 0);
  adj_kernel<<<96, 64, 0, stream>>>(f2, A2);
  adjx  <<<cdiv(96LL * 2048, 256), 256, 0, stream>>>(A2, Xb2, y2, 96);
  gemm32<<<dim3(12, 2), 128, 0, stream>>>(y2, th2t, th2_b, bnM2, h2,
                                          768, 128, 256, 3);
  mean_scatter<<<cdiv(96LL * 512, 256), 256, 0, stream>>>(h2, feats, 96, 3, 2560);

  // ---- FC head (M=32, 16x16 tiles) ----
  gemm16<<<dim3(2, 4), 128, 0, stream>>>(feats, fc1t, fc1_b, hf1, 32, 256, 4096);
  gemm16<<<dim3(2, 4), 128, 0, stream>>>(hf1, fc2t, fc2_b, hf2, 32, 256, 256);
  gemm16<<<dim3(2, 2), 128, 0, stream>>>(hf2, fc3t, fc3_b, hf3, 32, 128, 256);
  fc4_kernel<<<1, 160, 0, stream>>>(hf3, fc4_w, fc4_b, (float*)d_out);
}
